// _SpanMarkerV0_69303592288789
// MI455X (gfx1250) — matrix operand order repaired and checked by reading, not compiled, not run on hardware
//
#include <hip/hip_runtime.h>
#include <hip/hip_bf16.h>
#include <stdint.h>

// Problem constants (from reference): B=4, L=512, D=768, W=12, K=L*W=6144
#define PD   768
#define PL   512
#define PB   4
#define PK   6144           // spans per batch
#define NROW (PB * PK)      // 24576 span rows
#define NTOK (PB * PL)      // 2048 tokens

typedef __attribute__((ext_vector_type(16))) __bf16 v16bf;
typedef __attribute__((ext_vector_type(8)))  float  v8f;

// ---------------------------------------------------------------------------
// fp32 -> bf16 round-to-nearest-even
// ---------------------------------------------------------------------------
__device__ __forceinline__ unsigned short f32_to_bf16(float f) {
    union { float f; unsigned u; } c; c.f = f;
    unsigned u = c.u + 0x7FFFu + ((c.u >> 16) & 1u);
    return (unsigned short)(u >> 16);
}

__global__ void cvt_f32_to_bf16_kernel(const float* __restrict__ in,
                                       unsigned short* __restrict__ out,
                                       long n4) {
    long i = (long)blockIdx.x * blockDim.x + threadIdx.x;
    if (i >= n4) return;
    float4 v = ((const float4*)in)[i];
    ushort4 o;
    o.x = f32_to_bf16(v.x); o.y = f32_to_bf16(v.y);
    o.z = f32_to_bf16(v.z); o.w = f32_to_bf16(v.w);
    ((ushort4*)out)[i] = o;
}

// ---------------------------------------------------------------------------
// Gather span start/end rows, relu, concat -> cat[NROW][2*PD] (bf16)
// ---------------------------------------------------------------------------
__device__ __forceinline__ unsigned relu_pk_bf16(unsigned w) {
    unsigned lo = w & 0xFFFFu; if (lo & 0x8000u) lo = 0u;
    unsigned hi = w >> 16;     if (hi & 0x8000u) hi = 0u;
    return lo | (hi << 16);
}

__global__ void gather_cat_kernel(const unsigned short* __restrict__ srep,
                                  const unsigned short* __restrict__ erep,
                                  const long long* __restrict__ span,
                                  unsigned short* __restrict__ cat) {
    int row = blockIdx.x;                 // 0 .. NROW-1
    int b   = row / PK;
    long long s = span[(long long)row * 2 + 0];
    long long e = span[(long long)row * 2 + 1];
    int col = threadIdx.x * 8;            // 192 threads * 8 = 1536
    const unsigned short* src;
    if (col < PD) src = srep + ((long long)b * PL + s) * PD + col;
    else          src = erep + ((long long)b * PL + e) * PD + (col - PD);
    uint4 w = *(const uint4*)src;
    w.x = relu_pk_bf16(w.x); w.y = relu_pk_bf16(w.y);
    w.z = relu_pk_bf16(w.z); w.w = relu_pk_bf16(w.w);
    *(uint4*)(cat + (long long)row * (2 * PD) + col) = w;
}

// ---------------------------------------------------------------------------
// Tiled bf16 WMMA GEMM:  C[M][N] = act(A[M][Kd] * W[N][Kd]^T + bias[N])
//   block tile 128x128, K-tile 64, 256 threads (8 wave32: 2(M) x 4(N))
//   double-buffered LDS filled by global_load_async_to_lds_b128 (ASYNCcnt),
//   each wave: 64x32 -> 4x2 tiles of v_wmma_f32_16x16x32_bf16, 16 WMMA/tile
// ---------------------------------------------------------------------------
#define TILE_M 128
#define TILE_N 128
#define TILE_K 64
#define LDSS   72                         // 64 + 8 bf16 pad (144B rows, 16B aligned)
#define BUFE   ((TILE_M + TILE_N) * LDSS) // elements per buffer

__global__ __launch_bounds__(256, 2)
void gemm_bf16_wmma_kernel(const unsigned short* __restrict__ A, int lda,
                           const unsigned short* __restrict__ W, int ldw,
                           const float* __restrict__ bias,
                           void* __restrict__ Cv, int ldc,
                           int Kd, int doRelu, int outBf16) {
    // Single static LDS allocation -> starts at LDS byte offset 0; raw byte
    // offsets into it are passed to the async-DMA instructions below while
    // compute reads the same memory through normal pointers. The smem pointer
    // is passed into the asm (unused) so its address ESCAPES: without this,
    // GlobalsAA proves the never-stored LDS array can't be written by the asm
    // and folds all fragment loads to undef (observed in round 2).
    __shared__ unsigned short smem[2 * BUFE];

    const int tid  = threadIdx.x;
    const int lane = tid & 31;
    const int wave = tid >> 5;      // 0..7
    const int wm   = wave & 1;      // 2 waves along M (64 rows each)
    const int wn   = wave >> 1;     // 4 waves along N (32 cols each)
    const int l16  = lane & 15;
    const int half = lane >> 4;     // 0 | 1

    const int rowBase = blockIdx.y * TILE_M;
    const int colBase = blockIdx.x * TILE_N;

    // Async-copy assignment: 2 threads per row, 32 bf16 (64B) per thread.
    const int ldRow = tid >> 1;           // 0..127
    const int ldCol = (tid & 1) * 32;     // 0 | 32
    const unsigned short* gA0 = A + (long long)(rowBase + ldRow) * lda + ldCol;
    const unsigned short* gW0 = W + (long long)(colBase + ldRow) * ldw + ldCol;
    const unsigned ldsAoff = (unsigned)((ldRow * LDSS + ldCol) * 2);
    const unsigned ldsBoff = (unsigned)((TILE_M * LDSS + ldRow * LDSS + ldCol) * 2);

    // Issue one K-tile's worth of async global->LDS DMA (8 x b128 per thread).
    // IOFFSET applies to both the LDS and memory address, so consecutive 16B
    // chunks of a row stream with a single base address per matrix.
    auto issue_tile = [&](int t, int p) {
        unsigned long long ga = (unsigned long long)(gA0 + t * TILE_K);
        unsigned long long gw = (unsigned long long)(gW0 + t * TILE_K);
        unsigned la = ldsAoff + (unsigned)(p * BUFE * 2);
        unsigned lb = ldsBoff + (unsigned)(p * BUFE * 2);
        asm volatile(
            "global_load_async_to_lds_b128 %0, %2, off\n\t"
            "global_load_async_to_lds_b128 %0, %2, off offset:16\n\t"
            "global_load_async_to_lds_b128 %0, %2, off offset:32\n\t"
            "global_load_async_to_lds_b128 %0, %2, off offset:48\n\t"
            "global_load_async_to_lds_b128 %1, %3, off\n\t"
            "global_load_async_to_lds_b128 %1, %3, off offset:16\n\t"
            "global_load_async_to_lds_b128 %1, %3, off offset:32\n\t"
            "global_load_async_to_lds_b128 %1, %3, off offset:48"
            :: "v"(la), "v"(lb), "v"(ga), "v"(gw), "v"((const void*)smem)
            : "memory");
    };

    v8f acc[4][2];
    v8f vzero = {0.f,0.f,0.f,0.f,0.f,0.f,0.f,0.f};
#pragma unroll
    for (int i = 0; i < 4; ++i)
#pragma unroll
        for (int j = 0; j < 2; ++j) acc[i][j] = vzero;

    auto compute_tile = [&](int cur) {
        const unsigned short* Asb = smem + cur * BUFE;
        const unsigned short* Bsb = Asb + TILE_M * LDSS;
#pragma unroll
        for (int ks = 0; ks < 2; ++ks) {          // two 32-deep WMMA steps
            const int koff = ks * 32;
            union Frag { v16bf v; uint4 q[2]; } af[4], bfg[2];
            // A frag (16x32): lane<16 -> M=l16, K{0..7,16..23}; lane>=16 -> K{8..15,24..31}
#pragma unroll
            for (int ti = 0; ti < 4; ++ti) {
                const unsigned short* p =
                    &Asb[(wm * 64 + ti * 16 + l16) * LDSS + koff + half * 8];
                af[ti].q[0] = *(const uint4*)(p);
                af[ti].q[1] = *(const uint4*)(p + 16);
            }
            // B frag (32x16): lane<16 -> N=l16, K{0..7,16..23}; lane>=16 -> K{8..15,24..31}
#pragma unroll
            for (int tj = 0; tj < 2; ++tj) {
                const unsigned short* p =
                    &Bsb[(wn * 32 + tj * 16 + l16) * LDSS + koff + half * 8];
                bfg[tj].q[0] = *(const uint4*)(p);
                bfg[tj].q[1] = *(const uint4*)(p + 16);
            }
#pragma unroll
            for (int ti = 0; ti < 4; ++ti)
#pragma unroll
                for (int tj = 0; tj < 2; ++tj)
                    acc[ti][tj] = __builtin_amdgcn_wmma_f32_16x16x32_bf16(
                        false, af[ti].v, false, bfg[tj].v,
                        (short)0, acc[ti][tj], false, false);
        }
    };

    const int nk = Kd / TILE_K;
    issue_tile(0, 0);
    if (nk > 1) issue_tile(1, 1);

    // Steady state: 16 async ops outstanding; wait<=8 retires the tile we are
    // about to consume (async loads complete in order). Branch-free immediates.
    int kt = 0;
    for (; kt + 2 < nk; ++kt) {
        asm volatile("s_wait_asynccnt 0x8" ::: "memory");
        __syncthreads();
        compute_tile(kt & 1);
        __syncthreads();                  // all waves done reading buffer
        issue_tile(kt + 2, kt & 1);
    }
    if (nk > 1) {                         // penultimate tile (next still in flight)
        asm volatile("s_wait_asynccnt 0x8" ::: "memory");
        __syncthreads();
        compute_tile(kt & 1);
        ++kt;
    }
    asm volatile("s_wait_asynccnt 0x0" ::: "memory");   // last tile
    __syncthreads();
    compute_tile(kt & 1);

    // Epilogue: C tile layout — VGPR v: lanes0-15 M=v, lanes16-31 M=v+8; N=lane&15
    float*          Cf = (float*)Cv;
    unsigned short* Ch = (unsigned short*)Cv;
#pragma unroll
    for (int tj = 0; tj < 2; ++tj) {
        const int gcol = colBase + wn * 32 + tj * 16 + l16;
        const float bv = bias[gcol];
#pragma unroll
        for (int ti = 0; ti < 4; ++ti) {
            const int grow0 = rowBase + wm * 64 + ti * 16 + half * 8;
            v8f a = acc[ti][tj];
#pragma unroll
            for (int v = 0; v < 8; ++v) {
                float x = a[v] + bv;
                if (doRelu) x = x > 0.f ? x : 0.f;
                long long idx = (long long)(grow0 + v) * ldc + gcol;
                if (outBf16) Ch[idx] = f32_to_bf16(x);
                else         Cf[idx] = x;
            }
        }
    }
}

// ---------------------------------------------------------------------------
// Host orchestration
// ---------------------------------------------------------------------------
extern "C" void kernel_launch(void* const* d_in, const int* in_sizes, int n_in,
                              void* d_out, int out_size, void* d_ws, size_t ws_size,
                              hipStream_t stream) {
    (void)in_sizes; (void)n_in; (void)out_size; (void)ws_size;

    const float*     h    = (const float*)d_in[0];
    const long long* span = (const long long*)d_in[1];
    const float* ws1 = (const float*)d_in[2];
    const float* bs1 = (const float*)d_in[3];
    const float* ws2 = (const float*)d_in[4];
    const float* bs2 = (const float*)d_in[5];
    const float* we1 = (const float*)d_in[6];
    const float* be1 = (const float*)d_in[7];
    const float* we2 = (const float*)d_in[8];
    const float* be2 = (const float*)d_in[9];
    const float* wo1 = (const float*)d_in[10];
    const float* bo1 = (const float*)d_in[11];
    const float* wo2 = (const float*)d_in[12];
    const float* bo2 = (const float*)d_in[13];

    char* wsb = (char*)d_ws;
    size_t off = 0;
    auto alloc = [&](long long elems) -> unsigned short* {
        unsigned short* p = (unsigned short*)(wsb + off);
        off += ((size_t)elems * 2 + 255) & ~(size_t)255;
        return p;
    };
    unsigned short* hbf   = alloc((long long)NTOK * PD);        // 2048 x 768
    unsigned short* w1sb  = alloc(3072LL * 768);                // ws1 bf16
    unsigned short* w1eb  = alloc(3072LL * 768);                // we1 bf16
    unsigned short* w2sb  = alloc(768LL * 3072);                // ws2 bf16
    unsigned short* w2eb  = alloc(768LL * 3072);                // we2 bf16
    unsigned short* wo1b  = alloc(3072LL * 1536);               // wo1 bf16
    unsigned short* wo2b  = alloc(768LL * 3072);                // wo2 bf16
    unsigned short* H1    = alloc((long long)NTOK * 6144);      // stacked hidden
    unsigned short* srep  = alloc((long long)NTOK * PD);
    unsigned short* erep  = alloc((long long)NTOK * PD);
    unsigned short* cat   = alloc((long long)NROW * 2 * PD);    // 24576 x 1536
    unsigned short* H2    = alloc((long long)NROW * 3072);      // 24576 x 3072

    auto cvt = [&](const float* src, unsigned short* dst, long n) {
        long n4 = n / 4;
        unsigned blocks = (unsigned)((n4 + 255) / 256);
        cvt_f32_to_bf16_kernel<<<blocks, 256, 0, stream>>>(src, dst, n4);
    };
    auto gemm = [&](const unsigned short* A, int lda,
                    const unsigned short* Wm, int ldw, const float* bias,
                    void* C, int ldc, int M, int N, int Kd, int relu, int obf) {
        dim3 grid(N / TILE_N, M / TILE_M);
        gemm_bf16_wmma_kernel<<<grid, 256, 0, stream>>>(A, lda, Wm, ldw, bias,
                                                        C, ldc, Kd, relu, obf);
    };

    // 1) fp32 -> bf16 conversions
    cvt(h,   hbf,  (long)NTOK * PD);
    cvt(ws1, w1sb, 3072L * 768);
    cvt(we1, w1eb, 3072L * 768);
    cvt(ws2, w2sb, 768L * 3072);
    cvt(we2, w2eb, 768L * 3072);
    cvt(wo1, wo1b, 3072L * 1536);
    cvt(wo2, wo2b, 768L * 3072);

    // 2) start/end first layers -> stacked H1 (cols 0..3071 | 3072..6143)
    gemm(hbf, PD, w1sb, PD, bs1, H1,        6144, NTOK, 3072, PD, 1, 1);
    gemm(hbf, PD, w1eb, PD, be1, H1 + 3072, 6144, NTOK, 3072, PD, 1, 1);

    // 3) second layers -> start_rep / end_rep (bf16, no relu)
    gemm(H1,        6144, w2sb, 3072, bs2, srep, PD, NTOK, PD, 3072, 0, 1);
    gemm(H1 + 3072, 6144, w2eb, 3072, be2, erep, PD, NTOK, PD, 3072, 0, 1);

    // 4) gather + relu + concat
    gather_cat_kernel<<<NROW, 192, 0, stream>>>(srep, erep, span, cat);

    // 5) output MLP (dominant cost)
    gemm(cat, 2 * PD, wo1b, 2 * PD, bo1, H2, 3072, NROW, 3072, 2 * PD, 1, 1);
    gemm(H2, 3072, wo2b, 3072, bo2, d_out, PD, NROW, PD, 3072, 0, 0);
}
